// MySeq2SeqEncoder_19713899889143
// MI455X (gfx1250) — matrix operand order repaired and checked by reading, not compiled, not run on hardware
//
#include <hip/hip_runtime.h>
#include <hip/hip_bf16.h>

// Problem constants (match reference)
#define B_ 64
#define S_ 512
#define E_ 256
#define H_ 512
#define G_ 1536   // 3*H

typedef unsigned short u16;
typedef __attribute__((ext_vector_type(16))) __bf16 v16bf;
typedef __attribute__((ext_vector_type(8)))  float  v8f;
typedef __attribute__((ext_vector_type(4)))  unsigned int u32x4;
typedef __attribute__((ext_vector_type(8)))  int i32x8;
typedef __attribute__((ext_vector_type(4)))  int i32x4;

union ABTile { uint4 q[2]; v16bf v; };

__device__ __forceinline__ u16 f2bf(float f) {
  unsigned u = __float_as_uint(f);
  u += 0x7FFFu + ((u >> 16) & 1u);      // round-to-nearest-even
  return (u16)(u >> 16);
}

// ---------------------------------------------------------------------------
// TDM: 2D tile load global -> LDS with per-row LDS padding done in hardware.
//   tile = tile_rows x dwords_per_row (data_size = 4B), row stride = row length
//   pad_interval/pad_amount encoded per D# group1 (interval 2^(c+1) DWORDs,
//   amount (c+1) DWORDs) -> produces the bank-rotated LDS row strides.
// Descriptor packing per CDNA5 ISA.
// This toolchain exposes the 6-arg builtin: (g0, g1, g2, g3, g4, cpol).
// ---------------------------------------------------------------------------
__device__ __forceinline__ void tdm_load_2d(unsigned lds_addr, unsigned long long ga,
                                            unsigned tile_rows, unsigned dwords_per_row,
                                            unsigned pad_interval_code, unsigned pad_amount_code,
                                            unsigned tensor_rows) {
  u32x4 g0;
  g0[0] = 1u;                                                   // count=1 user descriptor
  g0[1] = lds_addr;                                             // LDS byte address
  g0[2] = (unsigned)ga;                                         // global_addr[31:0]
  g0[3] = (unsigned)((ga >> 32) & 0x01FFFFFFu) | (2u << 30);    // global_addr[56:32] | type=2
  i32x8 g1;
  g1[0] = (int)((2u << 16) | (1u << 20) |                      // data_size=4B, pad_enable=1
                (pad_interval_code << 22) | (pad_amount_code << 25));
  g1[1] = (int)((dwords_per_row & 0xFFFFu) << 16);              // tensor_dim0[15:0]
  g1[2] = (int)((dwords_per_row >> 16) |                        // tensor_dim0[31:16]
                ((tensor_rows & 0xFFFFu) << 16));               // tensor_dim1[15:0]
  g1[3] = (int)((tensor_rows >> 16) |                           // tensor_dim1[31:16]
                (dwords_per_row << 16));                        // tile_dim0
  g1[4] = (int)tile_rows;                                       // tile_dim1 | tile_dim2(=0)<<16
  g1[5] = (int)dwords_per_row;                                  // tensor_dim0_stride[31:0]
  g1[6] = 0;                                                    // stride0[47:32] | stride1[15:0]
  g1[7] = 0;                                                    // stride1[47:16]
  i32x4 gz4 = {0, 0, 0, 0};
  i32x8 gz8 = {0, 0, 0, 0, 0, 0, 0, 0};
  __builtin_amdgcn_tensor_load_to_lds(g0, g1, gz4, gz4, gz8, 0);
}

// ---------------------------------------------------------------------------
// Prep kernel 1: gather embeddings -> time-major bf16 ex[s][b][e]
// ---------------------------------------------------------------------------
__global__ void k_embed(const int* __restrict__ x, const float* __restrict__ emb,
                        u16* __restrict__ exb) {
  size_t v = (size_t)blockIdx.x * blockDim.x + threadIdx.x;   // vector-of-8 index
  const size_t NV = (size_t)S_ * B_ * E_ / 8;
  if (v >= NV) return;
  size_t e8 = (v * 8) % E_;
  size_t sb = (v * 8) / E_;            // s*B + b
  int s = (int)(sb / B_);
  int b = (int)(sb % B_);
  int tok = x[(size_t)b * S_ + s];
  const float* src = emb + (size_t)tok * E_ + e8;
  u16 o[8];
#pragma unroll
  for (int i = 0; i < 8; ++i) o[i] = f2bf(src[i]);
  *(uint4*)(exb + v * 8) = *(const uint4*)o;
}

// ---------------------------------------------------------------------------
// Prep kernel 2: f32 -> bf16 weight conversion
// ---------------------------------------------------------------------------
__global__ void k_cvt(const float* __restrict__ src, u16* __restrict__ dst, int n8) {
  int v = blockIdx.x * blockDim.x + threadIdx.x;
  if (v >= n8) return;
  const float* s = src + (size_t)v * 8;
  u16 o[8];
#pragma unroll
  for (int i = 0; i < 8; ++i) o[i] = f2bf(s[i]);
  *(uint4*)(dst + (size_t)v * 8) = *(const uint4*)o;
}

// ---------------------------------------------------------------------------
// Prep kernel 3: zero h buffer 0 (f32 + bf16) and compute lengths[b]
// ---------------------------------------------------------------------------
__global__ void k_init(const int* __restrict__ x, float* __restrict__ hf,
                       u16* __restrict__ hb, int* __restrict__ len) {
  int i = blockIdx.x * blockDim.x + threadIdx.x;
  if (i < B_ * H_) { hf[i] = 0.0f; hb[i] = 0; }
  if (i < B_) {
    int c = 0;
    for (int s = 0; s < S_; ++s) c += (x[(size_t)i * S_ + s] != 0);
    len[i] = c;
  }
}

// LDS staging geometry: padded row strides to rotate banks (528B / 1040B rows)
#define WIH_STR 264                 // u16 units  (256 + 8 pad) = 132 DWORDs
#define WHH_STR 520                 // u16 units  (512 + 8 pad) = 260 DWORDs
#define LDS_WHH_OFF (48 * WIH_STR)  // u16 offset of W_hh region
#define LDS_U16_TOTAL (48 * WIH_STR + 48 * WHH_STR)

// ---------------------------------------------------------------------------
// GRU step: one launch per timestep t.
// Grid: 32 blocks (jb = 16-wide hidden-column slice), 128 threads = 4 waves,
// wave w owns batch tile mb = w. All 4 waves share the same B tiles: the
// weight slice (3 gates x 16 rows of W_ih and W_hh) is staged into LDS by
// SIX Tensor-Data-Mover descriptors issued from wave 0 (TENSORcnt), with the
// bank-rotation padding inserted by the TDM pad engine. B tiles then come
// from LDS (ds_load_b128), A tiles stream from global into WMMA.
// ---------------------------------------------------------------------------
__global__ __launch_bounds__(128)
void k_gru_step(const u16* __restrict__ exb, const u16* __restrict__ wihb,
                const u16* __restrict__ whhb, const float* __restrict__ bih,
                const float* __restrict__ bhh, float* __restrict__ hf,
                u16* __restrict__ hb, const int* __restrict__ len,
                float* __restrict__ out, int t) {
  __shared__ __align__(16) u16 lds_w[LDS_U16_TOTAL];

  const int lane = threadIdx.x & 31;
  const int mb   = threadIdx.x >> 5;     // 0..3  : batch tile
  const int jb   = blockIdx.x;           // 0..31 : hidden-column tile

  // ---- stage weight slice into LDS via TDM (wave 0 only; EXEC-independent) ----
  const unsigned lds_base = (unsigned)(size_t)&lds_w[0];
  if ((threadIdx.x >> 5) == 0) {
#pragma unroll
    for (int gate = 0; gate < 3; ++gate) {
      // W_ih slice: 16 rows x 128 DWORDs, pad 4 DWORDs every 128 -> stride 132 DW
      tdm_load_2d(lds_base + (unsigned)(gate * 16 * WIH_STR * 2),
                  (unsigned long long)(size_t)(wihb + (size_t)(gate * H_ + jb * 16) * E_),
                  16u, (unsigned)(E_ / 2), 6u, 3u, (unsigned)G_);
      // W_hh slice: 16 rows x 256 DWORDs, pad 4 DWORDs every 256 -> stride 260 DW
      tdm_load_2d(lds_base + (unsigned)((LDS_WHH_OFF + gate * 16 * WHH_STR) * 2),
                  (unsigned long long)(size_t)(whhb + (size_t)(gate * H_ + jb * 16) * H_),
                  16u, (unsigned)(H_ / 2), 7u, 3u, (unsigned)G_);
    }
    __builtin_amdgcn_s_wait_tensorcnt(0);
  }

  const int cur = t & 1, nxt = cur ^ 1;
  const float* hf_cur = hf + (size_t)cur * (B_ * H_);
  float*       hf_nxt = hf + (size_t)nxt * (B_ * H_);
  const u16*   hb_cur = hb + (size_t)cur * (B_ * H_);
  u16*         hb_nxt = hb + (size_t)nxt * (B_ * H_);

  const int n0 = lane & 15;                      // N index within tile
  const int gr = jb * 16, gz = H_ + jb * 16, gn = 2 * H_ + jb * 16;

  // accumulators initialized with biases broadcast along M
  const float br = bih[gr + n0] + bhh[gr + n0];
  const float bz = bih[gz + n0] + bhh[gz + n0];
  const float bi = bih[gn + n0];
  const float bh = bhh[gn + n0];
  v8f accr, accz, acci, acch;
#pragma unroll
  for (int i = 0; i < 8; ++i) { accr[i] = br; accz[i] = bz; acci[i] = bi; acch[i] = bh; }

  // Operand addressing used here (wave32; contiguous k, same order for A and B):
  // A 16x32 bf16: lane -> row m=lane&15; 16 contiguous k at aks (aks = 0|16)
  // B 32x16 bf16: lane -> col n=lane&15; 16 contiguous k at bks (bks = 0|16)
  const int am  = lane & 15;
  const int aks = (lane >> 4) << 4;   // 0 or 16
  const int bks = (lane >> 4) << 4;   // 0 or 16

  // LDS row bases for this lane's B column (row = gate*16 + n0)
  const int lwr = (0 * 16 + n0) * WIH_STR + bks;
  const int lwz = (1 * 16 + n0) * WIH_STR + bks;
  const int lwn = (2 * 16 + n0) * WIH_STR + bks;
  const int lvr = LDS_WHH_OFF + (0 * 16 + n0) * WHH_STR + bks;
  const int lvz = LDS_WHH_OFF + (1 * 16 + n0) * WHH_STR + bks;
  const int lvn = LDS_WHH_OFF + (2 * 16 + n0) * WHH_STR + bks;

  const u16* arow = exb + ((size_t)t * B_ + mb * 16 + am) * E_ + aks;
  const u16* hrow = hb_cur + (size_t)(mb * 16 + am) * H_ + aks;
  __builtin_prefetch(arow, 0, 3);
  __builtin_prefetch(hrow, 0, 3);

  // TDM writes visible to whole workgroup after the barrier
  __syncthreads();

  // ---- input projection part: K = E = 256 (A = ex[t] global, B = LDS) ----
  for (int kk = 0; kk < E_; kk += 32) {
    ABTile a, tr, tz, tn;
    a.q[0]  = *(const uint4*)(arow + kk);
    a.q[1]  = *(const uint4*)(arow + kk + 8);
    tr.q[0] = *(const uint4*)&lds_w[lwr + kk];  tr.q[1] = *(const uint4*)&lds_w[lwr + kk + 8];
    tz.q[0] = *(const uint4*)&lds_w[lwz + kk];  tz.q[1] = *(const uint4*)&lds_w[lwz + kk + 8];
    tn.q[0] = *(const uint4*)&lds_w[lwn + kk];  tn.q[1] = *(const uint4*)&lds_w[lwn + kk + 8];
    accr = __builtin_amdgcn_wmma_f32_16x16x32_bf16(false, a.v, false, tr.v, (short)0, accr, false, false);
    accz = __builtin_amdgcn_wmma_f32_16x16x32_bf16(false, a.v, false, tz.v, (short)0, accz, false, false);
    acci = __builtin_amdgcn_wmma_f32_16x16x32_bf16(false, a.v, false, tn.v, (short)0, acci, false, false);
  }

  // ---- hidden projection part: K = H = 512 (A = h_cur global, B = LDS) ----
  for (int kk = 0; kk < H_; kk += 32) {
    ABTile a, tr, tz, tn;
    a.q[0]  = *(const uint4*)(hrow + kk);
    a.q[1]  = *(const uint4*)(hrow + kk + 8);
    tr.q[0] = *(const uint4*)&lds_w[lvr + kk];  tr.q[1] = *(const uint4*)&lds_w[lvr + kk + 8];
    tz.q[0] = *(const uint4*)&lds_w[lvz + kk];  tz.q[1] = *(const uint4*)&lds_w[lvz + kk + 8];
    tn.q[0] = *(const uint4*)&lds_w[lvn + kk];  tn.q[1] = *(const uint4*)&lds_w[lvn + kk + 8];
    accr = __builtin_amdgcn_wmma_f32_16x16x32_bf16(false, a.v, false, tr.v, (short)0, accr, false, false);
    accz = __builtin_amdgcn_wmma_f32_16x16x32_bf16(false, a.v, false, tz.v, (short)0, accz, false, false);
    acch = __builtin_amdgcn_wmma_f32_16x16x32_bf16(false, a.v, false, tn.v, (short)0, acch, false, false);
  }

  // ---- gates + state update. C/D layout: VGPR v, lane<16 -> M=v, lane>=16 -> M=v+8
  const int j   = jb * 16 + n0;
  const int mhi = (lane >> 4) << 3;
#pragma unroll
  for (int v = 0; v < 8; ++v) {
    int b = mb * 16 + v + mhi;
    float r  = 1.0f / (1.0f + __expf(-accr[v]));
    float z  = 1.0f / (1.0f + __expf(-accz[v]));
    float n  = tanhf(acci[v] + r * acch[v]);
    float hp = hf_cur[(size_t)b * H_ + j];
    float hn = (1.0f - z) * n + z * hp;
    hf_nxt[(size_t)b * H_ + j] = hn;
    hb_nxt[(size_t)b * H_ + j] = f2bf(hn);
    if (t == len[b] - 1) out[(size_t)b * H_ + j] = hn;
  }
}

// ---------------------------------------------------------------------------
// Launch: prep (gather/convert/init) then 512 dependent step launches.
// Workspace layout (bytes, all regions 256B aligned), total ~18.6 MB:
//   exb   @ 0          : S*B*E bf16       = 16,777,216
//   wihb  @ 16777216   : 1536*256 bf16    =    786,432
//   whhb  @ 17563648   : 1536*512 bf16    =  1,572,864
//   hf    @ 19136512   : 2 * B*H f32     =    262,144
//   hb    @ 19398656   : 2 * B*H bf16    =    131,072
//   len   @ 19529728   : B int            =        256
// ---------------------------------------------------------------------------
extern "C" void kernel_launch(void* const* d_in, const int* in_sizes, int n_in,
                              void* d_out, int out_size, void* d_ws, size_t ws_size,
                              hipStream_t stream) {
  const int*   x   = (const int*)d_in[0];
  const float* emb = (const float*)d_in[1];
  const float* Wih = (const float*)d_in[2];
  const float* Whh = (const float*)d_in[3];
  const float* bih = (const float*)d_in[4];
  const float* bhh = (const float*)d_in[5];
  float* out = (float*)d_out;

  char* ws = (char*)d_ws;
  u16*   exb  = (u16*)(ws + 0);
  u16*   wihb = (u16*)(ws + 16777216);
  u16*   whhb = (u16*)(ws + 17563648);
  float* hf   = (float*)(ws + 19136512);
  u16*   hb   = (u16*)(ws + 19398656);
  int*   len  = (int*)(ws + 19529728);

  // prep
  k_embed<<<dim3((S_ * B_ * E_ / 8 + 255) / 256), dim3(256), 0, stream>>>(x, emb, exb);
  k_cvt<<<dim3((G_ * E_ / 8 + 255) / 256), dim3(256), 0, stream>>>(Wih, wihb, G_ * E_ / 8);
  k_cvt<<<dim3((G_ * H_ / 8 + 255) / 256), dim3(256), 0, stream>>>(Whh, whhb, G_ * H_ / 8);
  k_init<<<dim3((B_ * H_ + 255) / 256), dim3(256), 0, stream>>>(x, hf, hb, len);

  // recurrence: 512 dependent launches (stream-ordered, graph-friendly)
  for (int t = 0; t < S_; ++t) {
    k_gru_step<<<dim3(H_ / 16), dim3(128), 0, stream>>>(
        exb, wihb, whhb, bih, bhh, hf, hb, len, out, t);
  }
}
